// ActorGNNMLP_81655918232307
// MI455X (gfx1250) — matrix operand order repaired and checked by reading, not compiled, not run on hardware
//
#include <hip/hip_runtime.h>
#include <hip/hip_bf16.h>

// ---------------------------------------------------------------------------
// ActorGNNMLP for MI455X (gfx1250, wave32, WMMA).
//
// gconv(x,W) = diag(deg_out^-1/2) @ x @ W scattered to dst, scaled by
// deg_in^-1/2, plus bias.  Row scaling commutes with x@W, so:
//   Y = X @ W              (dense GEMM, bf16 WMMA, f32 accumulate)
//   agg[dst] += s_out[src] * Y[src]   (f32 global atomics, L2-resident)
//   h = relu( sum_rel agg*s_in + b )  (f32 combine)
// Weights are pre-transposed/converted to zero-padded bf16 [M][Kp] so the
// WMMA B-fragment is two contiguous 16B loads per lane with no guards.
// ---------------------------------------------------------------------------

typedef __bf16 bf16_t;
typedef bf16_t bf16x4 __attribute__((ext_vector_type(4)));
typedef bf16_t bf16x8 __attribute__((ext_vector_type(8)));
typedef bf16_t v16bf  __attribute__((ext_vector_type(16)));
typedef float  v8f    __attribute__((ext_vector_type(8)));

#define N_REGION 5000
#define N_DRIVER 5000
#define N_ORDER  50000
#define N_POI    20000
#define F_REGION 4
#define F_ORDER  6
#define F_POI    4
#define F_DRIVER 5025
#define HID      128
#define EMB      64
#define MLP_H    256
#define ACT      26

// ---------------------------------------------------------------------------
// Utility kernels
// ---------------------------------------------------------------------------
__global__ void agm_zero_kernel(float* __restrict__ p, long n) {
    long t = (long)blockIdx.x * blockDim.x + threadIdx.x;
    if (t < n) p[t] = 0.0f;
}

__global__ void agm_count_kernel(const int* __restrict__ idx, int ne,
                                 float* __restrict__ cnt) {
    int t = blockIdx.x * blockDim.x + threadIdx.x;
    if (t < ne) atomicAdd(&cnt[idx[t]], 1.0f);
}

__global__ void agm_rsqrt_kernel(float* __restrict__ c, int n) {
    int t = blockIdx.x * blockDim.x + threadIdx.x;
    if (t < n) c[t] = rsqrtf(fmaxf(c[t], 1.0f));
}

// ---------------------------------------------------------------------------
// Weight transpose+convert: Bt[m][k] = bf16(W[k][m]), K zero-padded to Kp.
// ---------------------------------------------------------------------------
__global__ void agm_wt_convert_kernel(const float* __restrict__ W,
                                      bf16_t* __restrict__ Bt,
                                      int K, int M, int Kp) {
    long t = (long)blockIdx.x * blockDim.x + threadIdx.x;
    if (t >= (long)M * Kp) return;
    int mcol = (int)(t / Kp);
    int k    = (int)(t % Kp);
    Bt[t] = (k < K) ? (bf16_t)W[(long)k * M + mcol] : (bf16_t)0.0f;
}

// ---------------------------------------------------------------------------
// Small-K dense GEMM (K = 4 or 6): Y[n x M] = X[n x K] @ W[K x M]
// ---------------------------------------------------------------------------
__global__ void agm_smallk_gemm_kernel(const float* __restrict__ X,
                                       const float* __restrict__ W,
                                       float* __restrict__ Y,
                                       int n, int K, int M) {
    long t = (long)blockIdx.x * blockDim.x + threadIdx.x;
    if (t >= (long)n * M) return;
    int r = (int)(t / M);
    int c = (int)(t % M);
    float acc = 0.0f;
    #pragma unroll 1
    for (int k = 0; k < K; ++k) acc += X[(long)r * K + k] * W[(long)k * M + c];
    Y[t] = acc;
}

// ---------------------------------------------------------------------------
// WMMA GEMM:  Y[n x M] = A[n x K] @ Bt^T   (+bias, +relu optional)
//   A  : f32 row-major n x K (staged -> bf16 LDS tile, shared by 4 waves)
//   Bt : bf16 row-major M x Kp (pre-transposed, zero-padded; no guards)
//   block = 128 threads (4 waves); block tile = 16 rows x 64 cols;
//   wave w owns a 16x16 column slab.  Fragments per CDNA5 ISA 7.12.2.
//   Interior K-tiles are guard-free (row index clamped: clamped rows only
//   feed D-rows that are never stored).  Single guarded tail tile for K%32.
// ---------------------------------------------------------------------------
__global__ __launch_bounds__(128)
void agm_wmma_gemm_kernel(const float* __restrict__ A,
                          const bf16_t* __restrict__ Bt,
                          const float* __restrict__ bias,
                          float* __restrict__ Y,
                          int nRows, int K, int Kp, int M, int doRelu) {
    __shared__ __align__(16) bf16_t As[16][32];

    const int tid  = threadIdx.x;
    const int lane = tid & 31;
    const int wave = tid >> 5;
    const int rowbase = blockIdx.x * 16;
    const int colbase = blockIdx.y * 64;
    const int m  = lane & 15;   // row (A frag) / col (B frag) within tile
    const int hi = lane >> 4;   // lane half selector

    // A staging assignment: thread -> (row, 4-col chunk); row CLAMPED not guarded
    const int arow  = tid >> 3;
    const int akoff = (tid & 7) * 4;
    int gr = rowbase + arow;
    if (gr >= nRows) gr = nRows - 1;
    const float* aptr = A + (long)gr * K + akoff;

    // B fragment source: lane's column row in Bt, two 8-wide K chunks
    const int bcol = colbase + wave * 16 + m;
    const bf16_t* brow = Bt + (long)bcol * Kp + 8 * hi;

    v8f acc = {0.f, 0.f, 0.f, 0.f, 0.f, 0.f, 0.f, 0.f};
    const int Kfull = K & ~31;

    for (int k0 = 0; k0 < Kfull; k0 += 32) {
        __syncthreads();
        if (k0 + 32 < K) __builtin_prefetch(aptr + k0 + 32, 0, 0);
        // guard-free staging: 4 consecutive floats -> packed bf16x4, 8B ds store
        const float* ap = aptr + k0;
        bf16x4 bv;
        bv[0] = (bf16_t)ap[0]; bv[1] = (bf16_t)ap[1];
        bv[2] = (bf16_t)ap[2]; bv[3] = (bf16_t)ap[3];
        *(bf16x4*)&As[arow][akoff] = bv;
        __syncthreads();

        // A fragment: lane L: M=L%16; elems 0..7 -> K=8*hi+e; 8..15 -> 16+8*hi+(e-8)
        v16bf a, b;
        {
            const bf16x8 a0 = *(const bf16x8*)&As[m][8 * hi];
            const bf16x8 a1 = *(const bf16x8*)&As[m][16 + 8 * hi];
            #pragma unroll
            for (int i = 0; i < 8; ++i) { a[i] = a0[i]; a[i + 8] = a1[i]; }
        }
        // B fragment: lane L: N=L%16; elems 0..7 -> K=8*hi+e; 8..15 -> 16+8*hi+(e-8)
        {
            const bf16x8 b0 = *(const bf16x8*)(brow + k0);
            const bf16x8 b1 = *(const bf16x8*)(brow + k0 + 16);
            #pragma unroll
            for (int i = 0; i < 8; ++i) { b[i] = b0[i]; b[i + 8] = b1[i]; }
        }
        acc = __builtin_amdgcn_wmma_f32_16x16x32_bf16(
            false, a, false, b, (short)0, acc, false, false);
    }

    if (Kfull < K) {    // single guarded tail tile (K % 32 != 0)
        __syncthreads();
        #pragma unroll
        for (int i = 0; i < 4; ++i) {
            int kk = akoff + i;
            int gk = Kfull + kk;
            As[arow][kk] = (gk < K) ? (bf16_t)A[(long)gr * K + gk] : (bf16_t)0.0f;
        }
        __syncthreads();
        v16bf a, b;
        {
            const bf16x8 a0 = *(const bf16x8*)&As[m][8 * hi];
            const bf16x8 a1 = *(const bf16x8*)&As[m][16 + 8 * hi];
            #pragma unroll
            for (int i = 0; i < 8; ++i) { a[i] = a0[i]; a[i + 8] = a1[i]; }
        }
        {   // Bt is zero-padded to Kp = Kfull+32: guard-free
            const bf16x8 b0 = *(const bf16x8*)(brow + Kfull);
            const bf16x8 b1 = *(const bf16x8*)(brow + Kfull + 16);
            #pragma unroll
            for (int i = 0; i < 8; ++i) { b[i] = b0[i]; b[i + 8] = b1[i]; }
        }
        acc = __builtin_amdgcn_wmma_f32_16x16x32_bf16(
            false, a, false, b, (short)0, acc, false, false);
    }

    // C/D layout: lanes 0-15 rows 0..7, lanes 16-31 rows 8..15; N = lane%16
    const int col = colbase + wave * 16 + m;
    #pragma unroll
    for (int v = 0; v < 8; ++v) {
        int r = rowbase + v + 8 * hi;
        if (r < nRows) {
            float o = acc[v];
            if (bias)   o += bias[col];
            if (doRelu) o = fmaxf(o, 0.0f);
            Y[(long)r * M + col] = o;
        }
    }
}

// ---------------------------------------------------------------------------
// Edge scatter: agg[dst[e]] += s_out[src[e]] * Y[src[e]]   (float4 granules)
// ---------------------------------------------------------------------------
__global__ void agm_scatter_kernel(const float* __restrict__ Y,
                                   const int* __restrict__ src,
                                   const int* __restrict__ dst,
                                   const float* __restrict__ sOut,
                                   float* __restrict__ agg,
                                   int ne, int mquads) {
    long t = (long)blockIdx.x * blockDim.x + threadIdx.x;
    if (t >= (long)ne * mquads) return;
    int e = (int)(t / mquads);
    int q = (int)(t % mquads);
    int s = src[e], d = dst[e];
    float sc = sOut[s];
    const float4 y = ((const float4*)Y)[(long)s * mquads + q];
    float* a = agg + ((long)d * mquads + q) * 4;
    atomicAdd(a + 0, sc * y.x);
    atomicAdd(a + 1, sc * y.y);
    atomicAdd(a + 2, sc * y.z);
    atomicAdd(a + 3, sc * y.w);
}

// ---------------------------------------------------------------------------
// Combine: out = [relu]( a0*si0 + b0 + a1*si1 + b1 [+ a2*si2+b2 + a3*si3+b3] )
// ---------------------------------------------------------------------------
__global__ void agm_combine2_kernel(const float* __restrict__ a0,
                                    const float* __restrict__ a1,
                                    const float* __restrict__ si0,
                                    const float* __restrict__ si1,
                                    const float* __restrict__ b0,
                                    const float* __restrict__ b1,
                                    float* __restrict__ out,
                                    int n, int M, int doRelu) {
    long t = (long)blockIdx.x * blockDim.x + threadIdx.x;
    if (t >= (long)n * M) return;
    int r = (int)(t / M), c = (int)(t % M);
    float v = a0[t] * si0[r] + b0[c] + a1[t] * si1[r] + b1[c];
    if (doRelu) v = fmaxf(v, 0.0f);
    out[t] = v;
}

__global__ void agm_combine4_kernel(const float* __restrict__ a0,
                                    const float* __restrict__ a1,
                                    const float* __restrict__ a2,
                                    const float* __restrict__ a3,
                                    const float* __restrict__ si0,
                                    const float* __restrict__ si1,
                                    const float* __restrict__ si2,
                                    const float* __restrict__ si3,
                                    const float* __restrict__ b0,
                                    const float* __restrict__ b1,
                                    const float* __restrict__ b2,
                                    const float* __restrict__ b3,
                                    float* __restrict__ out,
                                    int n, int M) {
    long t = (long)blockIdx.x * blockDim.x + threadIdx.x;
    if (t >= (long)n * M) return;
    int r = (int)(t / M), c = (int)(t % M);
    float v = a0[t] * si0[r] + b0[c] + a1[t] * si1[r] + b1[c]
            + a2[t] * si2[r] + b2[c] + a3[t] * si3[r] + b3[c];
    out[t] = fmaxf(v, 0.0f);  // layer-1 region always relu'd
}

// ---------------------------------------------------------------------------
// Final: logits = X[n x 256] @ Wf3[256 x 26] + bf3 ; prob = softmax(logits)
// ---------------------------------------------------------------------------
__global__ void agm_final_kernel(const float* __restrict__ X,
                                 const float* __restrict__ W,
                                 const float* __restrict__ b,
                                 float* __restrict__ prob, int n) {
    int r = blockIdx.x * blockDim.x + threadIdx.x;
    if (r >= n) return;
    float acc[ACT];
    #pragma unroll
    for (int c = 0; c < ACT; ++c) acc[c] = b[c];
    for (int k = 0; k < MLP_H; ++k) {
        float xv = X[(long)r * MLP_H + k];
        #pragma unroll
        for (int c = 0; c < ACT; ++c) acc[c] += xv * W[k * ACT + c];
    }
    float mx = acc[0];
    #pragma unroll
    for (int c = 1; c < ACT; ++c) mx = fmaxf(mx, acc[c]);
    float s = 0.0f;
    #pragma unroll
    for (int c = 0; c < ACT; ++c) { acc[c] = __expf(acc[c] - mx); s += acc[c]; }
    float inv = 1.0f / s;
    #pragma unroll
    for (int c = 0; c < ACT; ++c) prob[(long)r * ACT + c] = acc[c] * inv;
}

// ---------------------------------------------------------------------------
// Host launcher
// ---------------------------------------------------------------------------
static inline int cdiv(long a, long b) { return (int)((a + b - 1) / b); }

extern "C" void kernel_launch(void* const* d_in, const int* in_sizes, int n_in,
                              void* d_out, int out_size, void* d_ws, size_t ws_size,
                              hipStream_t stream) {
    (void)n_in; (void)out_size; (void)ws_size;
    const float* x_region = (const float*)d_in[0];
    const float* x_driver = (const float*)d_in[1];
    const float* x_order  = (const float*)d_in[2];
    const float* x_poi    = (const float*)d_in[3];
    const int* esrc[6] = { (const int*)d_in[4],  (const int*)d_in[6],
                           (const int*)d_in[8],  (const int*)d_in[10],
                           (const int*)d_in[12], (const int*)d_in[14] };
    const int* edst[6] = { (const int*)d_in[5],  (const int*)d_in[7],
                           (const int*)d_in[9],  (const int*)d_in[11],
                           (const int*)d_in[13], (const int*)d_in[15] };
    const int   ne[6]  = { in_sizes[4], in_sizes[6], in_sizes[8],
                           in_sizes[10], in_sizes[12], in_sizes[14] };
    // relations: 0=r2r 1=d2r 2=d2d 3=r2d 4=o2r 5=p2r
    const int nsrc[6] = { N_REGION, N_DRIVER, N_DRIVER, N_REGION, N_ORDER, N_POI };
    const int ndst[6] = { N_REGION, N_REGION, N_DRIVER, N_DRIVER, N_REGION, N_REGION };

    int p = 16;
    const float* W1[6]; const float* B1[6];
    for (int i = 0; i < 6; ++i) { W1[i] = (const float*)d_in[p++]; B1[i] = (const float*)d_in[p++]; }
    const float *W2[4], *B2[4], *W3[4], *B3[4];   // params order: rr, dr, dd, rd
    for (int i = 0; i < 4; ++i) {
        W2[i] = (const float*)d_in[p++]; B2[i] = (const float*)d_in[p++];
        W3[i] = (const float*)d_in[p++]; B3[i] = (const float*)d_in[p++];
    }
    const float* Wf1 = (const float*)d_in[p++]; const float* bf1 = (const float*)d_in[p++];
    const float* Wf2 = (const float*)d_in[p++]; const float* bf2 = (const float*)d_in[p++];
    const float* Wf3 = (const float*)d_in[p++]; const float* bf3 = (const float*)d_in[p++];

    float* prob    = (float*)d_out;                          // 5000 x 26
    float* gnn_out = (float*)d_out + (long)N_DRIVER * ACT;   // 5000 x 64

    // ---- workspace bump allocator (256B aligned)
    char* base = (char*)d_ws;
    size_t off = 0;
    auto bumpb = [&](long nbytes) -> void* {
        void* q = (void*)(base + off);
        off += (size_t)((nbytes + 255) / 256) * 256;
        return q;
    };
    auto bump = [&](long nfl) -> float* { return (float*)bumpb(nfl * 4); };

    float* s_out[6]; float* s_in[6];
    for (int i = 0; i < 6; ++i) s_out[i] = bump(nsrc[i]);
    for (int i = 0; i < 6; ++i) s_in[i]  = bump(ndst[i]);
    float*  y     = bump((long)N_ORDER * HID);     // largest Y buffer, reused
    bf16_t* wbuf  = (bf16_t*)bumpb((long)HID * 5056 * 2); // max M*Kp bf16 (128x5056)
    float* aggR0 = bump((long)N_REGION * HID);
    float* aggR1 = bump((long)N_REGION * HID);
    float* aggR2 = bump((long)N_REGION * HID);
    float* aggR3 = bump((long)N_REGION * HID);
    float* aggD0 = bump((long)N_DRIVER * HID);
    float* aggD1 = bump((long)N_DRIVER * HID);
    float* hr    = bump((long)N_REGION * HID);
    float* hd    = bump((long)N_DRIVER * HID);
    float* hr2   = bump((long)N_REGION * HID);
    float* hd2   = bump((long)N_DRIVER * HID);
    float* x1    = bump((long)N_DRIVER * MLP_H);
    float* x2    = bump((long)N_DRIVER * MLP_H);

    const int T = 256;
    auto zero = [&](float* ptr, long n) {
        agm_zero_kernel<<<cdiv(n, T), T, 0, stream>>>(ptr, n);
    };
    auto degrees = [&](int rel) {
        zero(s_out[rel], nsrc[rel]);
        zero(s_in[rel],  ndst[rel]);
        agm_count_kernel<<<cdiv(ne[rel], T), T, 0, stream>>>(esrc[rel], ne[rel], s_out[rel]);
        agm_count_kernel<<<cdiv(ne[rel], T), T, 0, stream>>>(edst[rel], ne[rel], s_in[rel]);
        agm_rsqrt_kernel<<<cdiv(nsrc[rel], T), T, 0, stream>>>(s_out[rel], nsrc[rel]);
        agm_rsqrt_kernel<<<cdiv(ndst[rel], T), T, 0, stream>>>(s_in[rel],  ndst[rel]);
    };
    auto wmma_gemm = [&](const float* A, const float* W, const float* bias,
                         float* Y, int n, int K, int M, int relu) {
        int Kp = (K + 31) & ~31;
        agm_wt_convert_kernel<<<cdiv((long)M * Kp, T), T, 0, stream>>>(W, wbuf, K, M, Kp);
        dim3 g(cdiv(n, 16), M / 64);
        agm_wmma_gemm_kernel<<<g, 128, 0, stream>>>(A, wbuf, bias, Y, n, K, Kp, M, relu);
    };
    auto smallk = [&](const float* X, const float* W, float* Y, int n, int K, int M) {
        agm_smallk_gemm_kernel<<<cdiv((long)n * M, T), T, 0, stream>>>(X, W, Y, n, K, M);
    };
    auto scatter = [&](const float* Y, int rel, float* agg, int M) {
        int mq = M / 4;
        agm_scatter_kernel<<<cdiv((long)ne[rel] * mq, T), T, 0, stream>>>(
            Y, esrc[rel], edst[rel], s_out[rel], agg, ne[rel], mq);
    };

    for (int rel = 0; rel < 6; ++rel) degrees(rel);

    // ================= Layer 1 =================
    zero(aggR0, (long)N_REGION * HID); zero(aggR1, (long)N_REGION * HID);
    zero(aggR2, (long)N_REGION * HID); zero(aggR3, (long)N_REGION * HID);
    zero(aggD0, (long)N_DRIVER * HID); zero(aggD1, (long)N_DRIVER * HID);

    smallk(x_region, W1[0], y, N_REGION, F_REGION, HID);                scatter(y, 0, aggR0, HID); // r2r
    wmma_gemm(x_driver, W1[1], nullptr, y, N_DRIVER, F_DRIVER, HID, 0); scatter(y, 1, aggR1, HID); // d2r
    smallk(x_order, W1[4], y, N_ORDER, F_ORDER, HID);                   scatter(y, 4, aggR2, HID); // o2r
    smallk(x_poi,   W1[5], y, N_POI,   F_POI,   HID);                   scatter(y, 5, aggR3, HID); // p2r
    wmma_gemm(x_driver, W1[2], nullptr, y, N_DRIVER, F_DRIVER, HID, 0); scatter(y, 2, aggD0, HID); // d2d
    smallk(x_region, W1[3], y, N_REGION, F_REGION, HID);                scatter(y, 3, aggD1, HID); // r2d

    agm_combine4_kernel<<<cdiv((long)N_REGION * HID, T), T, 0, stream>>>(
        aggR0, aggR1, aggR2, aggR3, s_in[0], s_in[1], s_in[4], s_in[5],
        B1[0], B1[1], B1[4], B1[5], hr, N_REGION, HID);
    agm_combine2_kernel<<<cdiv((long)N_DRIVER * HID, T), T, 0, stream>>>(
        aggD0, aggD1, s_in[2], s_in[3], B1[2], B1[3], hd, N_DRIVER, HID, 1);

    // ================= Layer 2 =================
    zero(aggR0, (long)N_REGION * HID); zero(aggR1, (long)N_REGION * HID);
    zero(aggD0, (long)N_DRIVER * HID); zero(aggD1, (long)N_DRIVER * HID);

    wmma_gemm(hr, W2[0], nullptr, y, N_REGION, HID, HID, 0); scatter(y, 0, aggR0, HID); // r2r
    wmma_gemm(hd, W2[1], nullptr, y, N_DRIVER, HID, HID, 0); scatter(y, 1, aggR1, HID); // d2r
    wmma_gemm(hd, W2[2], nullptr, y, N_DRIVER, HID, HID, 0); scatter(y, 2, aggD0, HID); // d2d
    wmma_gemm(hr, W2[3], nullptr, y, N_REGION, HID, HID, 0); scatter(y, 3, aggD1, HID); // r2d

    agm_combine2_kernel<<<cdiv((long)N_REGION * HID, T), T, 0, stream>>>(
        aggR0, aggR1, s_in[0], s_in[1], B2[0], B2[1], hr2, N_REGION, HID, 1);
    agm_combine2_kernel<<<cdiv((long)N_DRIVER * HID, T), T, 0, stream>>>(
        aggD0, aggD1, s_in[2], s_in[3], B2[2], B2[3], hd2, N_DRIVER, HID, 1);

    // ================= Layer 3 ================= (only driver output used)
    zero(aggD0, (long)N_DRIVER * EMB); zero(aggD1, (long)N_DRIVER * EMB);
    wmma_gemm(hd2, W3[2], nullptr, y, N_DRIVER, HID, EMB, 0); scatter(y, 2, aggD0, EMB); // d2d
    wmma_gemm(hr2, W3[3], nullptr, y, N_REGION, HID, EMB, 0); scatter(y, 3, aggD1, EMB); // r2d
    agm_combine2_kernel<<<cdiv((long)N_DRIVER * EMB, T), T, 0, stream>>>(
        aggD0, aggD1, s_in[2], s_in[3], B3[2], B3[3], gnn_out, N_DRIVER, EMB, 0);

    // ================= MLP + softmax =================
    wmma_gemm(gnn_out, Wf1, bf1, x1, N_DRIVER, EMB,   MLP_H, 1);
    wmma_gemm(x1,      Wf2, bf2, x2, N_DRIVER, MLP_H, MLP_H, 1);
    agm_final_kernel<<<cdiv(N_DRIVER, T), T, 0, stream>>>(x2, Wf3, bf3, prob, N_DRIVER);
}
